// VectorQuantizerEMA_87883620811551
// MI455X (gfx1250) — matrix operand order repaired and checked by reading, not compiled, not run on hardware
//
#include <hip/hip_runtime.h>
#include <hip/hip_bf16.h>

// ---------------------------------------------------------------------------
// VQ-VAE EMA vector quantizer for MI455X (gfx1250, wave32, WMMA).
// N=32768 points, K=4096 codes, D=256.
// Distance argmin fused into v_wmma_f32_16x16x32_bf16; B tiles staged through
// LDS with double-buffered global_load_async_to_lds_b128 (ASYNCcnt), and the
// LDS->VGPR B fragments ping-ponged so WMMAs never wait on dscnt==0.
// ---------------------------------------------------------------------------

#define N_PTS   32768
#define K_CODES 4096
#define DIM     256
#define KTILES  (K_CODES / 16)
#define LDS_ROW 264   // 256 halfs + 8 halfs pad (16B) -> distinct bank quads

__device__ constexpr float BETA            = 0.25f;
__device__ constexpr float DECAY           = 0.99f;
__device__ constexpr float USAGE_DECAY     = 0.99f;
__device__ constexpr float EPSV            = 1e-5f;
__device__ constexpr float RESET_THRESHOLD = 100.0f;

typedef __attribute__((ext_vector_type(16))) __bf16 bf16x16;
typedef __attribute__((ext_vector_type(8)))  __bf16 bf16x8;
typedef __attribute__((ext_vector_type(8)))  float  f32x8;

union BF16x16U { bf16x16 v; bf16x8 h[2]; };

// ---------------------------------------------------------------------------
// K0: zero scratch accumulators (dw, counts, used, scalars)
// ---------------------------------------------------------------------------
__global__ void zero_ws(float* __restrict__ dw, float* __restrict__ counts,
                        float* __restrict__ used, float* __restrict__ scalars,
                        int ndw) {
  int i = blockIdx.x * blockDim.x + threadIdx.x;
  int stride = gridDim.x * blockDim.x;
  for (int j = i; j < ndw; j += stride) dw[j] = 0.0f;
  if (i < K_CODES) { counts[i] = 0.0f; used[i] = 0.0f; }
  if (i < 4) scalars[i] = 0.0f;
}

// ---------------------------------------------------------------------------
// K1: f32 -> bf16 with scale (x is pre-scaled by -2 so WMMA yields -2*x.e)
// ---------------------------------------------------------------------------
__global__ void scale_to_bf16(const float* __restrict__ in,
                              __bf16* __restrict__ out, float scale, int n) {
  int i = blockIdx.x * blockDim.x + threadIdx.x;
  if (i < n) out[i] = (__bf16)(in[i] * scale);
}

// ---------------------------------------------------------------------------
// K2: codebook row -> bf16 copy + e2[k] = sum(e[k,:]^2); one block per row
// ---------------------------------------------------------------------------
__global__ __launch_bounds__(256)
void prep_codebook(const float* __restrict__ emb, __bf16* __restrict__ eb,
                   float* __restrict__ e2) {
  __shared__ float red[256];
  const int k = blockIdx.x;
  const int t = threadIdx.x;
  float v = emb[(size_t)k * DIM + t];
  eb[(size_t)k * DIM + t] = (__bf16)v;
  red[t] = v * v;
  __syncthreads();
  for (int s = 128; s > 0; s >>= 1) {
    if (t < s) red[t] += red[t + s];
    __syncthreads();
  }
  if (t == 0) e2[k] = red[0];
}

// ---------------------------------------------------------------------------
// K3/K7: fused distance + argmin.
//   score(row, col) = e2[col] + (-2x[row,:]) . e[col,:]   (x^2 term dropped)
//
// Block = 8 waves; each wave owns a 16-row A tile resident in VGPRs for the
// whole K loop. The 16-code B tile (8KB) is loaded ONCE per block into LDS
// via async global->LDS b128 copies (double buffered, ASYNCcnt). Inside the
// tile, B fragments are ping-ponged between two register buffers so each
// WMMA only waits on ds loads issued one WMMA earlier. e2 is prefetched one
// tile ahead. Running min/argmin kept in the 8-slot accumulator layout;
// final cross-lane reduce with __shfl_xor within 16-lane halves.
// ---------------------------------------------------------------------------
__global__ __launch_bounds__(256)
void vq_argmin_wmma(const __bf16* __restrict__ xb,   // [N,D] bf16, scaled -2
                    const __bf16* __restrict__ eb,   // [K,D] bf16
                    const float*  __restrict__ e2,   // [K]
                    int*   __restrict__ idx_out,     // [N] or nullptr
                    float* __restrict__ fidx_out,    // [N] or nullptr
                    float* __restrict__ used_out) {  // [K] or nullptr
  __shared__ __bf16 Btile[2][16 * LDS_ROW];

  const int tid  = threadIdx.x;
  const int lane = tid & 31;
  const int wave = tid >> 5;
  const int row0 = (blockIdx.x * 8 + wave) * 16;
  const int m    = lane & 15;
  const int hi   = lane >> 4;

  // Resident A operand: 16 rows x 256 K. Per ISA 16-bit A layout:
  // VGPRs 0-3: K = 32c + hi*8 + [0..7], VGPRs 4-7: K = 32c + 16 + hi*8 + [0..7]
  BF16x16U A[8];
  {
    const __bf16* arow = xb + (size_t)(row0 + m) * DIM;
#pragma unroll
    for (int c = 0; c < 8; ++c) {
      const __bf16* p = arow + 32 * c + hi * 8;
      A[c].h[0] = *(const bf16x8*)(p);
      A[c].h[1] = *(const bf16x8*)(p + 16);
    }
  }

  // Cooperative async B-tile fill: thread tid copies 32 bytes of code row
  // cr = tid/16 (segment cs). Offset applies to both global and LDS address.
  const int cr = tid >> 4;          // code row within tile (0..15)
  const int cs = (tid & 15) * 16;   // half offset within row (0..240)
  auto issue_copy = [&](int t, int buf) {
    const __bf16* g = eb + ((size_t)(t * 16 + cr)) * DIM + cs;
    unsigned l = (unsigned)(unsigned long long)&Btile[buf][cr * LDS_ROW + cs];
    asm volatile("global_load_async_to_lds_b128 %0, %1, off"
                 :: "v"(l), "v"(g) : "memory");
    asm volatile("global_load_async_to_lds_b128 %0, %1, off offset:16"
                 :: "v"(l), "v"(g) : "memory");
  };

  float mv[8];
  int   mi[8];
#pragma unroll
  for (int s = 0; s < 8; ++s) { mv[s] = 3.4e38f; mi[s] = 0; }

  // Prime buffer 0 and the e2 pipeline.
  issue_copy(0, 0);
  float e2v = e2[m];
  asm volatile("s_wait_asynccnt 0x0" ::: "memory");
  __syncthreads();

  for (int t = 0; t < KTILES; ++t) {
    const int cur = t & 1;
    if (t + 1 < KTILES) issue_copy(t + 1, cur ^ 1);  // overlap with compute

    // Prefetch next tile's e2 (consumed next iteration; latency hidden).
    const float e2n = (t + 1 < KTILES) ? e2[(t + 1) * 16 + m] : 0.0f;

    // B layout (32x16, 16-bit): lane n = lane%16, K = 32c + hi*8 + [0..7], 32c + 16 + hi*8 + [0..7]
    const __bf16* bbase = &Btile[cur][m * LDS_ROW + hi * 8];

    // Ping-pong B fragment registers: chunk c+1 loads issue before WMMA(c),
    // so each WMMA's ds wait covers loads a full WMMA old.
    BF16x16U Bp[2];
    Bp[0].h[0] = *(const bf16x8*)(bbase);
    Bp[0].h[1] = *(const bf16x8*)(bbase + 16);

    f32x8 acc;
#pragma unroll
    for (int s = 0; s < 8; ++s) acc[s] = e2v;

#pragma unroll
    for (int c = 0; c < 8; ++c) {
      if (c + 1 < 8) {
        Bp[(c + 1) & 1].h[0] = *(const bf16x8*)(bbase + 32 * (c + 1));
        Bp[(c + 1) & 1].h[1] = *(const bf16x8*)(bbase + 32 * (c + 1) + 16);
      }
      acc = __builtin_amdgcn_wmma_f32_16x16x32_bf16(
          /*neg_a=*/false, A[c].v, /*neg_b=*/false, Bp[c & 1].v,
          /*c_mod=*/(short)0, acc, /*reuse_a=*/false, /*reuse_b=*/false);
    }

    const int col = t * 16 + m;
#pragma unroll
    for (int s = 0; s < 8; ++s) {
      if (acc[s] < mv[s]) { mv[s] = acc[s]; mi[s] = col; }
    }
    e2v = e2n;

    // Drain our async fills of the other buffer, then block-wide handoff.
    asm volatile("s_wait_asynccnt 0x0" ::: "memory");
    __syncthreads();
  }

  // Cross-lane argmin within each 16-lane half (cols live across lanes;
  // rows live in slots + lane-half).
#pragma unroll
  for (int s = 0; s < 8; ++s) {
    float v = mv[s];
    int   i = mi[s];
#pragma unroll
    for (int msk = 8; msk >= 1; msk >>= 1) {
      float ov = __shfl_xor(v, msk, 32);
      int   oi = __shfl_xor(i, msk, 32);
      if (ov < v || (ov == v && oi < i)) { v = ov; i = oi; }
    }
    mv[s] = v; mi[s] = i;
  }

  if (m == 0) {  // lane 0 (rows 0..7) and lane 16 (rows 8..15)
#pragma unroll
    for (int s = 0; s < 8; ++s) {
      const int row = row0 + hi * 8 + s;
      if (idx_out)  idx_out[row]  = mi[s];
      if (fidx_out) fidx_out[row] = (float)mi[s];
      if (used_out) used_out[mi[s]] = 1.0f;  // idempotent store
    }
  }
}

// ---------------------------------------------------------------------------
// K4: quantized gather, commitment-loss partial, counts + dw segment sums.
// One block (256 threads) per x row.
// ---------------------------------------------------------------------------
__global__ __launch_bounds__(256)
void gather_scatter(const float* __restrict__ x, const float* __restrict__ emb,
                    const int* __restrict__ idx, float* __restrict__ quant,
                    float* __restrict__ counts, float* __restrict__ dw,
                    float* __restrict__ loss_sum) {
  __shared__ float red[256];
  const int row = blockIdx.x;
  const int t   = threadIdx.x;
  const int k   = idx[row];
  const float xv = x[(size_t)row * DIM + t];
  const float q  = emb[(size_t)k * DIM + t];
  quant[(size_t)row * DIM + t] = q;  // x + stopgrad(q - x) == q numerically
  atomicAdd(&dw[(size_t)k * DIM + t], xv);
  const float d = xv - q;
  red[t] = d * d;
  __syncthreads();
  for (int s = 128; s > 0; s >>= 1) {
    if (t < s) red[t] += red[t + s];
    __syncthreads();
  }
  if (t == 0) {
    atomicAdd(loss_sum, red[0]);
    atomicAdd(&counts[k], 1.0f);
  }
}

// ---------------------------------------------------------------------------
// K5: EMA updates for cluster size, ema_w, usage; accumulate n = sum(new_cs)
// ---------------------------------------------------------------------------
__global__ void ema_update(const float* __restrict__ ema_cs,
                           const float* __restrict__ ema_w,
                           const float* __restrict__ usage,
                           const float* __restrict__ counts,
                           const float* __restrict__ dw,
                           float* __restrict__ out_cs,
                           float* __restrict__ out_ema_w,
                           float* __restrict__ out_usage,
                           float* __restrict__ n_sum,
                           float* __restrict__ usage_cnt) {
  const int i = blockIdx.x * blockDim.x + threadIdx.x;
  if (i < K_CODES * DIM)
    out_ema_w[i] = ema_w[i] * DECAY + (1.0f - DECAY) * dw[i];
  if (i < K_CODES) {
    const float cs = ema_cs[i] * DECAY + (1.0f - DECAY) * counts[i];
    out_cs[i] = cs;
    atomicAdd(n_sum, cs);
    const float u = usage[i] * USAGE_DECAY + (1.0f - USAGE_DECAY) * counts[i];
    out_usage[i] = u;
    if (u > 0.0f) atomicAdd(usage_cnt, 1.0f);
  }
}

// ---------------------------------------------------------------------------
// K6: normalized new embedding; also re-prime bf16 codebook + e2 for pass 2
// ---------------------------------------------------------------------------
__global__ __launch_bounds__(256)
void new_embedding_prep(const float* __restrict__ out_cs,
                        const float* __restrict__ out_ema_w,
                        const float* __restrict__ n_sum,
                        float* __restrict__ out_emb,
                        __bf16* __restrict__ eb,
                        float* __restrict__ e2) {
  __shared__ float red[256];
  const int k = blockIdx.x;
  const int t = threadIdx.x;
  const float n = *n_sum;
  const float cluster = (out_cs[k] + EPSV) / (n + (float)K_CODES * EPSV) * n;
  const float em = out_ema_w[(size_t)k * DIM + t] / cluster;
  out_emb[(size_t)k * DIM + t] = em;
  eb[(size_t)k * DIM + t] = (__bf16)em;
  red[t] = em * em;
  __syncthreads();
  for (int s = 128; s > 0; s >>= 1) {
    if (t < s) red[t] += red[t + s];
    __syncthreads();
  }
  if (t == 0) e2[k] = red[0];
}

// ---------------------------------------------------------------------------
// K8: dead-code reset. One block per code row.
// ---------------------------------------------------------------------------
__global__ __launch_bounds__(256)
void dead_reset(const float* __restrict__ used,
                const int* __restrict__ steps_in,
                const int* __restrict__ rand_idx,
                const float* __restrict__ x,
                float* __restrict__ out_emb,
                float* __restrict__ out_steps) {
  const int k = blockIdx.x;
  const int t = threadIdx.x;
  const float st = (used[k] > 0.0f) ? 0.0f : (float)steps_in[k] + 1.0f;
  const bool dead = st > RESET_THRESHOLD;
  if (dead) {
    const int r = rand_idx[k];
    out_emb[(size_t)k * DIM + t] = x[(size_t)r * DIM + t];
  }
  if (t == 0) out_steps[k] = dead ? 0.0f : st;
}

// ---------------------------------------------------------------------------
// K9: scalars
// ---------------------------------------------------------------------------
__global__ void finalize_scalars(const float* __restrict__ scalars,
                                 float* __restrict__ loss_out,
                                 float* __restrict__ rate_out) {
  *loss_out = BETA * scalars[1] / (float)((size_t)N_PTS * DIM);
  *rate_out = scalars[2] / (float)K_CODES;
}

// ---------------------------------------------------------------------------
// Host-side launch sequence (all on `stream`, graph-capture safe).
// ---------------------------------------------------------------------------
extern "C" void kernel_launch(void* const* d_in, const int* in_sizes, int n_in,
                              void* d_out, int out_size, void* d_ws, size_t ws_size,
                              hipStream_t stream) {
  (void)in_sizes; (void)n_in; (void)out_size; (void)ws_size;

  const float* x        = (const float*)d_in[0];  // [N,D]
  const float* emb      = (const float*)d_in[1];  // [K,D]
  const float* ema_cs   = (const float*)d_in[2];  // [K]
  const float* ema_w    = (const float*)d_in[3];  // [K,D]
  const float* usage    = (const float*)d_in[4];  // [K]
  const int*   steps_in = (const int*)d_in[5];    // [K]
  const int*   rand_idx = (const int*)d_in[6];    // [K]

  // Output layout (flat f32, reference return order)
  float* o = (float*)d_out;
  float* out_quant  = o;                                     // N*D
  float* out_idx    = out_quant + (size_t)N_PTS * DIM;       // N
  float* out_loss   = out_idx + N_PTS;                       // 1
  float* out_emb    = out_loss + 1;                          // K*D
  float* out_cs     = out_emb + (size_t)K_CODES * DIM;       // K
  float* out_ema_w  = out_cs + K_CODES;                      // K*D
  float* out_usage  = out_ema_w + (size_t)K_CODES * DIM;     // K
  float* out_steps  = out_usage + K_CODES;                   // K
  float* out_rate   = out_steps + K_CODES;                   // 1

  // Workspace layout (bytes)
  char* w = (char*)d_ws;
  __bf16* xb      = (__bf16*)(w);                                   // N*D*2
  __bf16* eb      = (__bf16*)(w + 16777216);                        // K*D*2
  float*  e2      = (float*)(w + 16777216 + 2097152);               // K
  int*    idx     = (int*)(w + 18890752);                           // N
  float*  counts  = (float*)(w + 19021824);                         // K
  float*  dw      = (float*)(w + 19038208);                         // K*D
  float*  used    = (float*)(w + 23232512);                         // K
  float*  scalars = (float*)(w + 23248896);                         // 4 (n, loss, usage_cnt)

  // K0: zero accumulators (4096*256 = K*D threads covers dw in one pass)
  zero_ws<<<4096, 256, 0, stream>>>(dw, counts, used, scalars, K_CODES * DIM);

  // K1: x -> bf16 scaled by -2 (folds the -2*x.e term into the WMMA)
  scale_to_bf16<<<(N_PTS * DIM + 255) / 256, 256, 0, stream>>>(x, xb, -2.0f, N_PTS * DIM);

  // K2: codebook -> bf16 + e2
  prep_codebook<<<K_CODES, 256, 0, stream>>>(emb, eb, e2);

  // K3: fused WMMA distance + argmin (pass 1)
  vq_argmin_wmma<<<N_PTS / 128, 256, 0, stream>>>(xb, eb, e2, idx, out_idx, nullptr);

  // K4: quantized gather + loss + segment sums
  gather_scatter<<<N_PTS, 256, 0, stream>>>(x, emb, idx, out_quant, counts, dw,
                                            scalars + 1);

  // K5: EMA updates + n accumulation
  ema_update<<<(K_CODES * DIM + 255) / 256, 256, 0, stream>>>(
      ema_cs, ema_w, usage, counts, dw, out_cs, out_ema_w, out_usage,
      scalars + 0, scalars + 2);

  // K6: normalized embedding; re-prime bf16 codebook + e2 for pass 2
  new_embedding_prep<<<K_CODES, 256, 0, stream>>>(out_cs, out_ema_w, scalars + 0,
                                                  out_emb, eb, e2);

  // K7: second argmin pass against updated codebook -> used flags
  vq_argmin_wmma<<<N_PTS / 128, 256, 0, stream>>>(xb, eb, e2, nullptr, nullptr, used);

  // K8: dead-code reset (overwrites dead rows of out_emb with x[rand_idx])
  dead_reset<<<K_CODES, 256, 0, stream>>>(used, steps_in, rand_idx, x, out_emb,
                                          out_steps);

  // K9: scalars
  finalize_scalars<<<1, 1, 0, stream>>>(scalars, out_loss, out_rate);
}
